// DreamOnPredictor_31396210934061
// MI455X (gfx1250) — matrix operand order repaired and checked by reading, not compiled, not run on hardware
//
#include <hip/hip_runtime.h>

typedef __attribute__((ext_vector_type(16))) __bf16 v16bf;
typedef __attribute__((ext_vector_type(8)))  float  v8f;
typedef __attribute__((ext_vector_type(4)))  unsigned int v4u;
typedef __attribute__((ext_vector_type(8)))  int    v8i;
typedef __attribute__((ext_vector_type(4)))  int    v4i;

#define VOCAB     32000
#define DMODEL    1024
#define NTOK      1024      // B*S = 2*512
#define MASK_ID   31999
#define EXPAND_ID 31998
#define STEPS     4
#define TOPK      50
#define TOPP      0.9f

// ---- Tensor Data Mover availability (gfx1250) ------------------------------
#if defined(__HIP_DEVICE_COMPILE__) && __has_builtin(__builtin_amdgcn_tensor_load_to_lds)
#define USE_TDM 1
#if __has_include(<hip/amd_detail/amd_gfx1250_TDM.h>)
#define TDM_6ARG 1          // therock/clang-23 toolchain: 6-arg builtin
#else
#define TDM_6ARG 0          // ROCm 7.2 / clang-22: 5-arg builtin
#endif
#else
#define USE_TDM 0
#endif

#if USE_TDM
// Issue one asynchronous 2-D tile load (global -> LDS) via the TDM.
// D# layout per CDNA5 ISA. data_size = 2 bytes.
// LDS padding: after every 16 dwords (one 32-elem bf16 row) skip 2 dwords,
// giving a 36-element (72 B) padded row pitch in LDS.
__device__ __forceinline__ void tdm_load_2d(unsigned lds_off, const void* gptr,
                                            unsigned tensor_d0, unsigned tensor_d1,
                                            unsigned tile_d0, unsigned tile_d1,
                                            unsigned long long stride0) {
  unsigned long long ga = (unsigned long long)(uintptr_t)gptr;
  v4u g0;
  g0.x = 1u;                                          // count=1 (valid, user mode)
  g0.y = lds_off;                                     // lds_addr (bytes)
  g0.z = (unsigned)(ga & 0xFFFFFFFFu);                // global_addr[31:0]
  g0.w = (unsigned)((ga >> 32) & 0x01FFFFFFu) | (2u << 30);  // addr[56:32] | type=2
  v8i g1;
  g1[0] = (int)((1u << 16) |                          // data_size = 2 B
                (1u << 20) |                          // pad_enable
                (3u << 22) |                          // pad_interval: 16 dwords
                (1u << 25));                          // pad_amount: 2 dwords
  g1[1] = (int)((tensor_d0 & 0xFFFFu) << 16);         // tensor_dim0[15:0]
  g1[2] = (int)((tensor_d0 >> 16) | ((tensor_d1 & 0xFFFFu) << 16));
  g1[3] = (int)((tensor_d1 >> 16) | (tile_d0 << 16)); // tensor_dim1[31:16] | tile_dim0
  g1[4] = (int)(tile_d1 & 0xFFFFu);                   // tile_dim1 (tile_dim2 = 0)
  g1[5] = (int)(stride0 & 0xFFFFFFFFull);             // tensor_dim0_stride[31:0]
  g1[6] = (int)((stride0 >> 32) & 0xFFFFull);         // stride[47:32] | dim1_stride=0
  g1[7] = 0;
  v4i z4 = {0, 0, 0, 0};
#if TDM_6ARG
  v8i z8 = {0, 0, 0, 0, 0, 0, 0, 0};
  __builtin_amdgcn_tensor_load_to_lds(g0, g1, z4, z4, z8, 0);
#else
  __builtin_amdgcn_tensor_load_to_lds(g0, g1, z4, z4, 0);
#endif
}
#endif

// ---------------- small helpers ---------------------------------------------
__global__ void k_copy_x(const int* __restrict__ src, int* __restrict__ dst) {
  int i = blockIdx.x * blockDim.x + threadIdx.x;
  if (i < NTOK) dst[i] = src[i];
}

// Tiled f32 -> bf16 transpose: dst[N x K] = src[K x N]^T  (one-time weight prep)
__global__ __launch_bounds__(256) void k_cvt_transpose(const float* __restrict__ src,
                                                       __bf16* __restrict__ dst,
                                                       int K, int N) {
  __shared__ float tile[32][33];
  const int tx = threadIdx.x & 31, ty = threadIdx.x >> 5;   // 32 x 8
  const int kb = blockIdx.y * 32, nb = blockIdx.x * 32;
  for (int i = ty; i < 32; i += 8)
    tile[i][tx] = src[(size_t)(kb + i) * N + nb + tx];
  __syncthreads();
  for (int i = ty; i < 32; i += 8)
    dst[(size_t)(nb + i) * K + kb + tx] = (__bf16)tile[tx][i];
}

// ---------------- embedding gather + convert --------------------------------
__global__ void k_gather(const float* __restrict__ emb, const int* __restrict__ x,
                         __bf16* __restrict__ h) {
  int t = blockIdx.x;
  int tok = x[t];
  const float* r = emb + (size_t)tok * DMODEL;
  __bf16* o = h + (size_t)t * DMODEL;
  for (int d = threadIdx.x; d < DMODEL; d += blockDim.x) o[d] = (__bf16)r[d];
}

// ---------------- bf16 WMMA GEMM: C[MxN] = op(A[MxK] @ Bt[NxK]^T) -----------
// Workgroup tile 64 x 128; 8 waves (4 m-rows x 2 n-cols); each wave owns
// 16 rows x 64 cols = four 16x16 f32 accumulators sharing one A fragment.
// Tiles staged into padded LDS (36-element rows) by the TDM, double-buffered:
// issue k+1, compute k, s_wait_tensorcnt, one barrier per K-step.
template<bool TANH>
__global__ __launch_bounds__(256) void k_gemm(const __bf16* __restrict__ A,
                                              const __bf16* __restrict__ Bt,
                                              __bf16* __restrict__ C,
                                              int M, int N, int K) {
  __shared__ __bf16 lds_a[2][64][36];    // [buf][m][k]
  __shared__ __bf16 lds_b[2][128][36];   // [buf][n][k]
  const int tid  = threadIdx.x;
  const int wave = tid >> 5;
  const int lane = tid & 31;
  const int wm   = wave >> 1;            // 0..3
  const int wn   = wave & 1;             // 0..1
  const int m0   = blockIdx.x * 64;
  const int n0   = blockIdx.y * 128;
  const int half = lane >> 4;
  const int l16  = lane & 15;

  v8f acc[4] = {};

#if USE_TDM
  if (wave == 0) {
    tdm_load_2d((unsigned)(uintptr_t)&lds_a[0][0][0], A + (size_t)m0 * K,
                (unsigned)K, (unsigned)M, 32u, 64u, (unsigned long long)K);
    tdm_load_2d((unsigned)(uintptr_t)&lds_b[0][0][0], Bt + (size_t)n0 * K,
                (unsigned)K, (unsigned)N, 32u, 128u, (unsigned long long)K);
    __builtin_amdgcn_s_wait_tensorcnt(0);
  }
  __syncthreads();

  int cur = 0;
  for (int k0 = 0; k0 < K; k0 += 32) {
    if (wave == 0 && k0 + 32 < K) {      // overlap next tile DMA with compute
      tdm_load_2d((unsigned)(uintptr_t)&lds_a[cur ^ 1][0][0],
                  A + (size_t)m0 * K + k0 + 32,
                  (unsigned)K, (unsigned)M, 32u, 64u, (unsigned long long)K);
      tdm_load_2d((unsigned)(uintptr_t)&lds_b[cur ^ 1][0][0],
                  Bt + (size_t)n0 * K + k0 + 32,
                  (unsigned)K, (unsigned)N, 32u, 128u, (unsigned long long)K);
    }
    // A fragment (contiguous k order, same as B): VGPR j -> K=16*half+2j;
    // K-pairs are dword-contiguous in the padded row.
    union { v16bf v; unsigned u[8]; } af, bfr;
    const int mrow = 16 * wm + l16;
    #pragma unroll
    for (int j = 0; j < 8; ++j) {
      int kb = 16 * half + 2 * j;
      af.u[j] = *(const unsigned*)&lds_a[cur][mrow][kb];
    }
    #pragma unroll
    for (int sub = 0; sub < 4; ++sub) {
      const int nc = 64 * wn + 16 * sub + l16;
      #pragma unroll
      for (int j = 0; j < 8; ++j)
        bfr.u[j] = *(const unsigned*)&lds_b[cur][nc][16 * half + 2 * j];
      acc[sub] = __builtin_amdgcn_wmma_f32_16x16x32_bf16(
          false, af.v, false, bfr.v, (short)0, acc[sub], false, false);
    }
    if (wave == 0) __builtin_amdgcn_s_wait_tensorcnt(0);
    __syncthreads();
    cur ^= 1;
  }
#else
  // Fallback: manual dword staging (both operands K-contiguous), single buffer.
  for (int k0 = 0; k0 < K; k0 += 32) {
    #pragma unroll
    for (int i = 0; i < 4; ++i) {
      int idx = tid + 256 * i;           // 1024 dwords of A
      int r = idx >> 4, c = idx & 15;
      *(unsigned*)&lds_a[0][r][c * 2] =
          *(const unsigned*)(A + (size_t)(m0 + r) * K + k0 + c * 2);
    }
    #pragma unroll
    for (int i = 0; i < 8; ++i) {
      int idx = tid + 256 * i;           // 2048 dwords of Bt
      int r = idx >> 4, c = idx & 15;
      *(unsigned*)&lds_b[0][r][c * 2] =
          *(const unsigned*)(Bt + (size_t)(n0 + r) * K + k0 + c * 2);
    }
    if (k0 + 32 < K) {
      __builtin_prefetch(A + (size_t)(m0 + (tid >> 2)) * K + k0 + 32, 0, 1);
      __builtin_prefetch(Bt + (size_t)(n0 + (tid >> 1)) * K + k0 + 32, 0, 1);
    }
    __syncthreads();
    union { v16bf v; unsigned u[8]; } af, bfr;
    const int mrow = 16 * wm + l16;
    #pragma unroll
    for (int j = 0; j < 8; ++j) {
      int kb = 16 * half + 2 * j;
      af.u[j] = *(const unsigned*)&lds_a[0][mrow][kb];
    }
    #pragma unroll
    for (int sub = 0; sub < 4; ++sub) {
      const int nc = 64 * wn + 16 * sub + l16;
      #pragma unroll
      for (int j = 0; j < 8; ++j)
        bfr.u[j] = *(const unsigned*)&lds_b[0][nc][16 * half + 2 * j];
      acc[sub] = __builtin_amdgcn_wmma_f32_16x16x32_bf16(
          false, af.v, false, bfr.v, (short)0, acc[sub], false, false);
    }
    __syncthreads();
  }
#endif

  // C/D layout: VGPR r -> M = 8*half + r ; N = lane%16.
  #pragma unroll
  for (int sub = 0; sub < 4; ++sub) {
    #pragma unroll
    for (int r = 0; r < 8; ++r) {
      int m = m0 + 16 * wm + 8 * half + r;
      int n = n0 + 64 * wn + 16 * sub + l16;
      float v = acc[sub][r];
      if (TANH) v = tanhf(v);
      C[(size_t)m * N + n] = (__bf16)v;
    }
  }
}

// ---------------- per-row top-p/top-k softmax stats -------------------------
// One 512-thread workgroup per output row. The whole penalized row (32000
// floats = 128 KB) lives in dynamic LDS (CDNA5: up to 320 KB per workgroup).
// conf = 1 / sum_{kept} exp(l - lmax), kept = top-min(n_topp, 50).
__global__ __launch_bounds__(512) void k_rowstat(const __bf16* __restrict__ logitsB,
                                                 float* __restrict__ conf_row,
                                                 int* __restrict__ x0_row) {
  extern __shared__ float row[];   // VOCAB floats
  __shared__ float sv[512];
  __shared__ int   si[512];
  const int tid = threadIdx.x;
  const int t = blockIdx.x;
  const int b = t >> 9, s = t & 511;
  const int src = (b << 9) + (s == 0 ? 0 : s - 1);   // shift-right of logits
  const __bf16* L = logitsB + (size_t)src * VOCAB;

  for (int n = tid; n < VOCAB; n += 512) {
    float v = (float)L[n];
    if (n == EXPAND_ID) v -= 1e9f;   // expand-token penalty
    row[n] = v;
  }
  __syncthreads();

  // max + argmax (first-index tiebreak)
  float m = -__builtin_inff(); int arg = 0;
  for (int n = tid; n < VOCAB; n += 512) {
    float v = row[n];
    if (v > m) { m = v; arg = n; }
  }
  sv[tid] = m; si[tid] = arg;
  __syncthreads();
  for (int off = 256; off > 0; off >>= 1) {
    if (tid < off) {
      float v2 = sv[tid + off]; int a2 = si[tid + off];
      if (v2 > sv[tid] || (v2 == sv[tid] && a2 < si[tid])) { sv[tid] = v2; si[tid] = a2; }
    }
    __syncthreads();
  }
  m = sv[0]; arg = si[0];
  __syncthreads();

  // full softmax denominator
  float z = 0.f;
  for (int n = tid; n < VOCAB; n += 512) z += __expf(row[n] - m);
  sv[tid] = z;
  __syncthreads();
  for (int off = 256; off > 0; off >>= 1) {
    if (tid < off) sv[tid] += sv[tid + off];
    __syncthreads();
  }
  const float Z = sv[0];
  __syncthreads();

  // successive-max selection of up to TOPK values, nucleus cut at TOPP*Z
  float cand = -__builtin_inff();
  for (int n = tid; n < VOCAB; n += 512) { float v = row[n]; if (v > cand) cand = v; }

  float csum = 0.f;
  for (int j = 0; j < TOPK; ++j) {
    sv[tid] = cand;
    __syncthreads();
    for (int off = 256; off > 0; off >>= 1) {
      if (tid < off) { float v2 = sv[tid + off]; if (v2 > sv[tid]) sv[tid] = v2; }
      __syncthreads();
    }
    float g = sv[0];
    __syncthreads();
    if (g == -__builtin_inff()) break;            // row exhausted
    if (j > 0 && csum > TOPP * Z) break;          // nucleus boundary
    csum += __expf(g - m);
    if (cand == g) {                              // owner advances its iterator
      float nc = -__builtin_inff();
      for (int n = tid; n < VOCAB; n += 512) {
        float v = row[n];
        if (v < g && v > nc) nc = v;
      }
      cand = nc;
    }
  }

  if (tid == 0) { conf_row[t] = 1.0f / csum; x0_row[t] = arg; }
}

// ---------------- global masked argmax + single-token transfer --------------
__global__ __launch_bounds__(1024) void k_finalize(int* __restrict__ x_cur,
                                                   const float* __restrict__ conf_row,
                                                   const int* __restrict__ x0_row,
                                                   float* __restrict__ confs, int step) {
  __shared__ float sv[1024];
  __shared__ int   si[1024];
  int t = threadIdx.x;
  float v = (x_cur[t] == MASK_ID) ? conf_row[t] : -__builtin_inff();
  sv[t] = v; si[t] = t;
  __syncthreads();
  for (int off = 512; off > 0; off >>= 1) {
    if (t < off) {
      float v2 = sv[t + off]; int a2 = si[t + off];
      if (v2 > sv[t] || (v2 == sv[t] && a2 < si[t])) { sv[t] = v2; si[t] = a2; }
    }
    __syncthreads();
  }
  if (t == 0) {
    float best = sv[0]; int idx = si[0];
    confs[step] = best;                           // -inf when no mask remains
    if (best > -__builtin_inff()) x_cur[idx] = x0_row[idx];
  }
}

// ---------------- output writeback ------------------------------------------
__global__ void k_writeback(const int* __restrict__ x_cur,
                            const float* __restrict__ confs,
                            float* __restrict__ out) {
  int t = blockIdx.x * blockDim.x + threadIdx.x;
  if (t < NTOK)              ((int*)out)[t] = x_cur[t];   // int32 token bits
  else if (t < NTOK + STEPS) out[t] = confs[t - NTOK];    // per-step confidences
}

// ---------------- host launcher ---------------------------------------------
extern "C" void kernel_launch(void* const* d_in, const int* in_sizes, int n_in,
                              void* d_out, int out_size, void* d_ws, size_t ws_size,
                              hipStream_t stream) {
  (void)in_sizes; (void)n_in; (void)out_size; (void)ws_size;
  const int*   x    = (const int*)d_in[0];
  const float* emb  = (const float*)d_in[1];
  const float* W1   = (const float*)d_in[2];
  const float* Wout = (const float*)d_in[3];

  char* ws = (char*)d_ws;
  size_t off = 0;
  auto alloc = [&](size_t bytes) -> void* {
    off = (off + 255) & ~(size_t)255;
    void* p = ws + off;
    off += bytes;
    return p;
  };
  __bf16* W1T   = (__bf16*)alloc((size_t)DMODEL * DMODEL * 2);  // [D x D]^T
  __bf16* WoutT = (__bf16*)alloc((size_t)VOCAB * DMODEL * 2);   // [V x D]
  __bf16* hb    = (__bf16*)alloc((size_t)NTOK * DMODEL * 2);
  __bf16* h1b   = (__bf16*)alloc((size_t)NTOK * DMODEL * 2);
  __bf16* logb  = (__bf16*)alloc((size_t)NTOK * VOCAB * 2);
  int*    xcur  = (int*)alloc((size_t)NTOK * 4);
  float*  confr = (float*)alloc((size_t)NTOK * 4);
  int*    x0r   = (int*)alloc((size_t)NTOK * 4);
  float*  confs = (float*)alloc((size_t)STEPS * 4);

  // one-time transposed bf16 weight prep (weights then live in MI455X L2)
  k_cvt_transpose<<<dim3(DMODEL / 32, DMODEL / 32), 256, 0, stream>>>(W1, W1T, DMODEL, DMODEL);
  k_cvt_transpose<<<dim3(VOCAB / 32, DMODEL / 32), 256, 0, stream>>>(Wout, WoutT, DMODEL, VOCAB);
  k_copy_x<<<4, 256, 0, stream>>>(x, xcur);

  for (int step = 0; step < STEPS; ++step) {
    k_gather<<<NTOK, 256, 0, stream>>>(emb, xcur, hb);
    k_gemm<true ><<<dim3(NTOK / 64, DMODEL / 128), 256, 0, stream>>>(
        hb, W1T, h1b, NTOK, DMODEL, DMODEL);
    k_gemm<false><<<dim3(NTOK / 64, VOCAB / 128), 256, 0, stream>>>(
        h1b, WoutT, logb, NTOK, VOCAB, DMODEL);
    k_rowstat<<<NTOK, 512, (size_t)VOCAB * sizeof(float), stream>>>(logb, confr, x0r);
    k_finalize<<<1, 1024, 0, stream>>>(xcur, confr, x0r, confs, step);
  }
  k_writeback<<<5, 256, 0, stream>>>(xcur, confs, (float*)d_out);
}
